// TransformerEncoder_47476568490302
// MI455X (gfx1250) — matrix operand order repaired and checked by reading, not compiled, not run on hardware
//
#include <hip/hip_runtime.h>
#include <cmath>

// ---------------------------------------------------------------------------
// Types for CDNA5 WMMA (wave32): v_wmma_f32_16x16x32_bf16
// ---------------------------------------------------------------------------
typedef __attribute__((ext_vector_type(16))) __bf16 v16bf;
typedef __attribute__((ext_vector_type(8)))  float  v8f;

union FragA { uint4 q[2]; v16bf v; };

__device__ __forceinline__ unsigned short f2bf(float f) {
  unsigned int u = __float_as_uint(f);
  u += 0x7FFFu + ((u >> 16) & 1u);          // round-to-nearest-even
  return (unsigned short)(u >> 16);
}

__device__ __forceinline__ v8f wmma_bf16(const FragA& a, const FragA& b, v8f c) {
  return __builtin_amdgcn_wmma_f32_16x16x32_bf16(false, a.v, false, b.v,
                                                 (short)0, c, false, false);
}

__device__ __forceinline__ float gelu_exact(float x) {
  return 0.5f * x * (1.0f + erff(x * 0.70710678118654752f));
}

// ---------------------------------------------------------------------------
// DPP16 butterfly reductions within each 16-lane group (no LDS, pure VALU).
// ---------------------------------------------------------------------------
template <int CTRL>
__device__ __forceinline__ float dpp_mv(float x) {
  return __int_as_float(
      __builtin_amdgcn_update_dpp(0, __float_as_int(x), CTRL, 0xF, 0xF, false));
}
__device__ __forceinline__ float red_max16(float x) {
  x = fmaxf(x, dpp_mv<0xB1>(x));    // quad_perm xor1
  x = fmaxf(x, dpp_mv<0x4E>(x));    // quad_perm xor2
  x = fmaxf(x, dpp_mv<0x141>(x));   // row_half_mirror
  x = fmaxf(x, dpp_mv<0x140>(x));   // row_mirror
  return x;
}
__device__ __forceinline__ float red_sum16(float x) {
  x += dpp_mv<0xB1>(x);
  x += dpp_mv<0x4E>(x);
  x += dpp_mv<0x141>(x);
  x += dpp_mv<0x140>(x);
  return x;
}

// ---------------------------------------------------------------------------
// Weight transpose + fp32 -> bf16 cast:  WT[n*K+k] = bf16(W[k*N+n]), K = 2^lgK
// ---------------------------------------------------------------------------
__global__ __launch_bounds__(256) void transpose_cast_kernel(
    const float* __restrict__ W, unsigned short* __restrict__ WT, int lgK, int N) {
  size_t idx = (size_t)blockIdx.x * 256 + threadIdx.x;
  int n = (int)(idx >> lgK);
  if (n >= N) return;
  int k = (int)(idx & ((1u << lgK) - 1u));
  WT[idx] = f2bf(W[(size_t)k * N + n]);
}

// ---------------------------------------------------------------------------
// LayerNorm over F=1024, fp32 in -> bf16 out. One row per block, 256 threads.
// ---------------------------------------------------------------------------
__global__ __launch_bounds__(256) void ln_cast_kernel(
    const float* __restrict__ X, const float* __restrict__ g,
    const float* __restrict__ bta, unsigned short* __restrict__ Y) {
  const int row = blockIdx.x;
  const int tid = threadIdx.x;
  float4 v = ((const float4*)(X + (size_t)row * 1024))[tid];
  float s  = v.x + v.y + v.z + v.w;
  float ss = v.x * v.x + v.y * v.y + v.z * v.z + v.w * v.w;
  s  = red_sum16(s);  ss = red_sum16(ss);        // within 16-lane groups
  s  += __shfl_xor(s,  16, 32);                  // cross 16-boundary once
  ss += __shfl_xor(ss, 16, 32);
  __shared__ float sh[16];
  int wv = tid >> 5, ln = tid & 31;
  if (ln == 0) { sh[wv] = s; sh[8 + wv] = ss; }
  __syncthreads();
  if (tid == 0) {
    float ts = 0.f, tss = 0.f;
    for (int i = 0; i < 8; ++i) { ts += sh[i]; tss += sh[8 + i]; }
    sh[0] = ts; sh[8] = tss;
  }
  __syncthreads();
  float mu  = sh[0] * (1.0f / 1024.0f);
  float var = sh[8] * (1.0f / 1024.0f) - mu * mu;
  float rs  = rsqrtf(var + 1e-5f);
  int c = tid * 4;
  ushort4 o;
  o.x = f2bf((v.x - mu) * rs * g[c + 0] + bta[c + 0]);
  o.y = f2bf((v.y - mu) * rs * g[c + 1] + bta[c + 1]);
  o.z = f2bf((v.z - mu) * rs * g[c + 2] + bta[c + 2]);
  o.w = f2bf((v.w - mu) * rs * g[c + 3] + bta[c + 3]);
  ((ushort4*)(Y + (size_t)row * 1024))[tid] = o;
}

// ---------------------------------------------------------------------------
// bf16 WMMA GEMM:  C[M,N] = A[M,K] * BT[N,K]^T + bias, fused epilogues.
// Block = 256 thr (8 waves); block tile 128 rows x 256 cols.
// Wave tile 64x64: 4 A-frags x 4 B-frags -> 16 WMMAs per k-step at
// 16x16B loads (1 load/WMMA) => ~32 FLOP per L2 byte.
// ---------------------------------------------------------------------------
enum { M_BF16 = 0, M_QK = 1, M_VT = 2, M_F32RES = 3, M_GELU_BF16 = 4, M_GELU_RES_F32 = 5 };

template <int MODE>
__global__ __launch_bounds__(256) void gemm_bf16_kernel(
    const unsigned short* __restrict__ A, const unsigned short* __restrict__ BT,
    const float* __restrict__ bias, const float* __restrict__ resid,
    unsigned short* __restrict__ outB, float* __restrict__ outF,
    int N, int K) {
  const int lane = threadIdx.x & 31;
  const int wave = threadIdx.x >> 5;
  const int half = lane >> 4;
  const int l16  = lane & 15;
  const int rowTile = blockIdx.y * 128 + (wave >> 2) * 64;
  const int colTile = blockIdx.x * 256 + (wave & 3) * 64;
  const unsigned short* aBase = A + (size_t)(rowTile + l16) * K;

  v8f acc[4][4] = {};
  for (int k0 = 0; k0 < K; k0 += 32) {
    FragA a[4];   // 16x32 bf16 A fragments (contiguous k = 16*half + s), rows r*16
#pragma unroll
    for (int r = 0; r < 4; ++r) {
      const unsigned short* ap = aBase + (size_t)(r * 16) * K + k0;
      a[r].q[0] = *(const uint4*)(ap + half * 16);
      a[r].q[1] = *(const uint4*)(ap + half * 16 + 8);
    }
#pragma unroll
    for (int t = 0; t < 4; ++t) {
      const unsigned short* bp =
          BT + (size_t)(colTile + t * 16 + l16) * K + k0 + half * 16;
      FragA b; // 32x16 bf16 B fragment: lane = N column, 16 contiguous K
      b.q[0] = *(const uint4*)(bp);
      b.q[1] = *(const uint4*)(bp + 8);
#pragma unroll
      for (int r = 0; r < 4; ++r) acc[r][t] = wmma_bf16(a[r], b, acc[r][t]);
    }
  }

#pragma unroll
  for (int r = 0; r < 4; ++r) {
#pragma unroll
    for (int t = 0; t < 4; ++t) {
#pragma unroll
      for (int i = 0; i < 8; ++i) {
        int row = rowTile + r * 16 + i + half * 8;   // C layout
        int col = colTile + t * 16 + l16;
        float v = acc[r][t][i] + bias[col];
        if constexpr (MODE == M_BF16) {
          outB[(size_t)row * N + col] = f2bf(v);
        } else if constexpr (MODE == M_QK) {
          int bb = row >> 11, n = row & 2047, hh = col >> 6, dh = col & 63;
          outB[((size_t)(bb * 16 + hh) * 2048 + n) * 64 + dh] = f2bf(v);
        } else if constexpr (MODE == M_VT) {
          int bb = row >> 11, n = row & 2047, hh = col >> 6, dh = col & 63;
          outB[((size_t)(bb * 16 + hh) * 64 + dh) * 2048 + n] = f2bf(v);
        } else if constexpr (MODE == M_F32RES) {
          outF[(size_t)row * N + col] = v + resid[(size_t)row * N + col];
        } else if constexpr (MODE == M_GELU_BF16) {
          outB[(size_t)row * N + col] = f2bf(gelu_exact(v));
        } else { // M_GELU_RES_F32
          outF[(size_t)row * N + col] = gelu_exact(v) + resid[(size_t)row * N + col];
        }
      }
    }
  }
}

// ---------------------------------------------------------------------------
// Flash attention with off-by-one softmax + per-head slimming.
// grid = (n/64, B*H), block = 128 (4 waves). Wave owns 16 Q rows and streams
// 2048 keys in chunks of 64 (4 score tiles per softmax update).
// Row statistics reduced with DPP16 butterflies.
// Q/K: [b,h,n,64] bf16, V: [b,h,64,n] bf16 (transposed). Out: [b,n,1024] bf16.
// ---------------------------------------------------------------------------
__global__ __launch_bounds__(128) void attn_kernel(
    const unsigned short* __restrict__ Qb, const unsigned short* __restrict__ Kb,
    const unsigned short* __restrict__ Vt, const float* __restrict__ slim,
    unsigned short* __restrict__ attnOut) {
  const int lane = threadIdx.x & 31;
  const int wave = threadIdx.x >> 5;
  const int half = lane >> 4;
  const int l16  = lane & 15;
  const int bh = blockIdx.y;
  const int h  = bh & 15;
  const int b  = bh >> 4;
  const int q0 = blockIdx.x * 64 + wave * 16;

  __shared__ __align__(16) unsigned short pb[4][16][64];

  const unsigned short* Qrow = Qb + ((size_t)bh * 2048 + q0 + l16) * 64;

  v8f o[4] = {};
  float mrow[8], srow[8];
#pragma unroll
  for (int i = 0; i < 8; ++i) { mrow[i] = -1e30f; srow[i] = 0.0f; }

  for (int kc = 0; kc < 2048; kc += 64) {
    // ---- scores: S = (Q . K^T) / 8 for 64 keys (four 16-wide N tiles) ----
    v8f s[4] = {};
#pragma unroll
    for (int kk = 0; kk < 64; kk += 32) {
      FragA a;
      a.q[0] = *(const uint4*)(Qrow + kk + half * 16);
      a.q[1] = *(const uint4*)(Qrow + kk + half * 16 + 8);
#pragma unroll
      for (int j = 0; j < 4; ++j) {
        const unsigned short* kp =
            Kb + ((size_t)bh * 2048 + kc + j * 16 + l16) * 64 + kk + half * 16;
        FragA bk;
        bk.q[0] = *(const uint4*)(kp); bk.q[1] = *(const uint4*)(kp + 8);
        s[j] = wmma_bf16(a, bk, s[j]);
      }
    }
    // ---- online off-by-one softmax update (rows = i + 8*half) ----
#pragma unroll
    for (int i = 0; i < 8; ++i) {
      float v0 = s[0][i] * 0.125f, v1 = s[1][i] * 0.125f;
      float v2 = s[2][i] * 0.125f, v3 = s[3][i] * 0.125f;
      float mx = red_max16(fmaxf(fmaxf(v0, v1), fmaxf(v2, v3)));
      float mnew = fmaxf(mrow[i], mx);
      float sc = __expf(mrow[i] - mnew);
      float p0 = __expf(v0 - mnew), p1 = __expf(v1 - mnew);
      float p2 = __expf(v2 - mnew), p3 = __expf(v3 - mnew);
      float ls = red_sum16((p0 + p1) + (p2 + p3));
      srow[i] = srow[i] * sc + ls;
      mrow[i] = mnew;
      o[0][i] *= sc; o[1][i] *= sc; o[2][i] *= sc; o[3][i] *= sc;
      pb[wave][i + half * 8][l16]      = f2bf(p0);
      pb[wave][i + half * 8][16 + l16] = f2bf(p1);
      pb[wave][i + half * 8][32 + l16] = f2bf(p2);
      pb[wave][i + half * 8][48 + l16] = f2bf(p3);
    }
    asm volatile("s_wait_dscnt 0" ::: "memory");   // LDS store -> load handoff
    // ---- O += P @ V : two 16x32 P fragments, V^T columns ----
    FragA ap0, ap1;
    ap0.q[0] = *(const uint4*)(&pb[wave][l16][half * 16]);
    ap0.q[1] = *(const uint4*)(&pb[wave][l16][half * 16 + 8]);
    ap1.q[0] = *(const uint4*)(&pb[wave][l16][32 + half * 16]);
    ap1.q[1] = *(const uint4*)(&pb[wave][l16][40 + half * 16]);
#pragma unroll
    for (int t = 0; t < 4; ++t) {
      const unsigned short* vp =
          Vt + ((size_t)bh * 64 + t * 16 + l16) * 2048 + kc + half * 16;
      FragA bv0, bv1;
      bv0.q[0] = *(const uint4*)(vp);      bv0.q[1] = *(const uint4*)(vp + 8);
      bv1.q[0] = *(const uint4*)(vp + 32); bv1.q[1] = *(const uint4*)(vp + 40);
      o[t] = wmma_bf16(ap0, bv0, o[t]);
      o[t] = wmma_bf16(ap1, bv1, o[t]);
    }
  }

  const float sl = slim[h];
#pragma unroll
  for (int i = 0; i < 8; ++i) {
    float sc = sl / (1.0f + srow[i]);     // off-by-one denominator
    int qrow = q0 + i + half * 8;
    unsigned short* orow = attnOut + ((size_t)b * 2048 + qrow) * 1024 + h * 64;
#pragma unroll
    for (int t = 0; t < 4; ++t) orow[t * 16 + l16] = f2bf(o[t][i] * sc);
  }
}

// ---------------------------------------------------------------------------
// Host orchestration
// ---------------------------------------------------------------------------
extern "C" void kernel_launch(void* const* d_in, const int* in_sizes, int n_in,
                              void* d_out, int out_size, void* d_ws, size_t ws_size,
                              hipStream_t stream) {
  (void)in_sizes; (void)n_in; (void)out_size; (void)ws_size;
  const float* x     = (const float*)d_in[0];
  const float* ln1_g = (const float*)d_in[1];
  const float* ln1_b = (const float*)d_in[2];
  const float* Wq    = (const float*)d_in[3];
  const float* bq    = (const float*)d_in[4];
  const float* Wk    = (const float*)d_in[5];
  const float* bk    = (const float*)d_in[6];
  const float* Wv    = (const float*)d_in[7];
  const float* bv    = (const float*)d_in[8];
  const float* Wo    = (const float*)d_in[9];
  const float* bo    = (const float*)d_in[10];
  const float* slim  = (const float*)d_in[11];
  const float* ln2_g = (const float*)d_in[12];
  const float* ln2_b = (const float*)d_in[13];
  const float* W1    = (const float*)d_in[14];
  const float* b1    = (const float*)d_in[15];
  const float* W2    = (const float*)d_in[16];
  const float* b2    = (const float*)d_in[17];

  const int F = 1024, MLP = 4096, TOK = 4096; // TOK = 2*2048
  char* ws = (char*)d_ws;
  size_t off = 0;
  auto carve = [&](size_t bytes) -> char* {
    char* p = ws + off; off += (bytes + 255) & ~(size_t)255; return p;
  };
  unsigned short* WqT  = (unsigned short*)carve((size_t)F * F * 2);
  unsigned short* WkT  = (unsigned short*)carve((size_t)F * F * 2);
  unsigned short* WvT  = (unsigned short*)carve((size_t)F * F * 2);
  unsigned short* WoT  = (unsigned short*)carve((size_t)F * F * 2);
  unsigned short* W1T  = (unsigned short*)carve((size_t)F * MLP * 2);
  unsigned short* W2T  = (unsigned short*)carve((size_t)F * MLP * 2);
  unsigned short* hB   = (unsigned short*)carve((size_t)TOK * F * 2);
  unsigned short* QbB  = (unsigned short*)carve((size_t)TOK * F * 2);
  unsigned short* KbB  = (unsigned short*)carve((size_t)TOK * F * 2);
  unsigned short* VtB  = (unsigned short*)carve((size_t)TOK * F * 2);
  unsigned short* attB = (unsigned short*)carve((size_t)TOK * F * 2);
  unsigned short* h2B  = (unsigned short*)carve((size_t)TOK * F * 2);
  unsigned short* m1B  = (unsigned short*)carve((size_t)TOK * MLP * 2);
  float*          outF = (float*)carve((size_t)TOK * F * 4);

  // 1) weights -> bf16 transposed (WT[n][k]); lgK = log2(K)
  transpose_cast_kernel<<<dim3((F * F) / 256), 256, 0, stream>>>(Wq, WqT, 10, F);
  transpose_cast_kernel<<<dim3((F * F) / 256), 256, 0, stream>>>(Wk, WkT, 10, F);
  transpose_cast_kernel<<<dim3((F * F) / 256), 256, 0, stream>>>(Wv, WvT, 10, F);
  transpose_cast_kernel<<<dim3((F * F) / 256), 256, 0, stream>>>(Wo, WoT, 10, F);
  transpose_cast_kernel<<<dim3((F * MLP) / 256), 256, 0, stream>>>(W1, W1T, 10, MLP);
  transpose_cast_kernel<<<dim3((F * MLP) / 256), 256, 0, stream>>>(W2, W2T, 12, F);

  // 2) LN1 -> bf16 activations
  ln_cast_kernel<<<dim3(TOK), 256, 0, stream>>>(x, ln1_g, ln1_b, hB);

  // 3) Q/K/V projections (scattered into attention-friendly layouts)
  gemm_bf16_kernel<M_QK><<<dim3(F / 256, TOK / 128), 256, 0, stream>>>(
      hB, WqT, bq, nullptr, QbB, nullptr, F, F);
  gemm_bf16_kernel<M_QK><<<dim3(F / 256, TOK / 128), 256, 0, stream>>>(
      hB, WkT, bk, nullptr, KbB, nullptr, F, F);
  gemm_bf16_kernel<M_VT><<<dim3(F / 256, TOK / 128), 256, 0, stream>>>(
      hB, WvT, bv, nullptr, VtB, nullptr, F, F);

  // 4) attention (off-by-one softmax + slimming), out = [b,n,F] bf16
  attn_kernel<<<dim3(2048 / 64, 32), 128, 0, stream>>>(QbB, KbB, VtB, slim, attB);

  // 5) output projection + residual-1 (fp32)
  gemm_bf16_kernel<M_F32RES><<<dim3(F / 256, TOK / 128), 256, 0, stream>>>(
      attB, WoT, bo, x, nullptr, outF, F, F);

  // 6) LN2 -> bf16
  ln_cast_kernel<<<dim3(TOK), 256, 0, stream>>>(outF, ln2_g, ln2_b, h2B);

  // 7) MLP up + GELU -> bf16
  gemm_bf16_kernel<M_GELU_BF16><<<dim3(MLP / 256, TOK / 128), 256, 0, stream>>>(
      h2B, W1T, b1, nullptr, m1B, nullptr, MLP, F);

  // 8) MLP down + GELU + residual-2 -> fp32 final output
  gemm_bf16_kernel<M_GELU_RES_F32><<<dim3(F / 256, TOK / 128), 256, 0, stream>>>(
      m1B, W2T, b2, outF, nullptr, (float*)d_out, F, MLP);
}
